// MixBlock_85495618994636
// MI455X (gfx1250) — matrix operand order repaired and checked by reading, not compiled, not run on hardware
//
#include <hip/hip_runtime.h>

// ---------------------------------------------------------------------------
// Types for CDNA5 WMMA (wave32). bf16 16x16x32, f32 accumulate.
// ---------------------------------------------------------------------------
typedef __bf16 bf16_t;
typedef __attribute__((ext_vector_type(16))) __bf16 v16bf;
typedef __attribute__((ext_vector_type(8)))  __bf16 v8bf;
typedef __attribute__((ext_vector_type(8)))  float  v8f;
typedef __attribute__((ext_vector_type(4)))  int    v4i;

#define EPSC 1e-5f
// softmax scale folded with log2(e): scores computed in exp2 domain
#define QSCALE (0.35355339059327373f * 1.4426950408889634f)

// Problem constants
#define NSP   8192   // T*H*W = 8*32*32
#define CIN   1024
#define CDIM  128
#define CC    64     // DIM_CONV == DIM_SA
#define ND    1024   // 4*16*16 downsampled tokens
#define QKV3  192

#ifndef __has_builtin
#define __has_builtin(x) 0
#endif
#if __has_builtin(__builtin_amdgcn_global_load_async_to_lds_b128) && \
    __has_builtin(__builtin_amdgcn_s_wait_asynccnt)
#define USE_ASYNC_LDS 1
#else
#define USE_ASYNC_LDS 0
#endif

// ---------------------------------------------------------------------------
// Fragment loaders (row-major, K contiguous per row).
// A 16-bit 16x32 (ISA 7.12.2): lane<16 holds K {k0..+7, k0+16..+23}; lane>=16
// holds {k0+8..+15, k0+24..+31}: two 16B chunks at +hi*8 and +hi*8+16.
// B 16-bit 32x16: same K order as A (GEMM loads B via load_frag_a at +hi*8);
// load_frag_b (unused) reads 16 consecutive bf16 (two 16B chunks).
// ---------------------------------------------------------------------------
__device__ inline v16bf load_frag_a(const bf16_t* __restrict__ p) {
  v8bf lo = *(const v8bf*)p;
  v8bf hi = *(const v8bf*)(p + 16);
  v16bf r;
#pragma unroll
  for (int i = 0; i < 8; ++i) { r[i] = lo[i]; r[i + 8] = hi[i]; }
  return r;
}
__device__ inline v16bf load_frag_b(const bf16_t* __restrict__ p) {
  v8bf lo = *(const v8bf*)p;
  v8bf hi = *(const v8bf*)(p + 8);
  v16bf r;
#pragma unroll
  for (int i = 0; i < 8; ++i) { r[i] = lo[i]; r[i + 8] = hi[i]; }
  return r;
}

// ---------------------------------------------------------------------------
// GEMM: C[N][M] (f32) = B[N][K](bf16) * A[M][K](bf16)^T + bias[M] (+Res)
// Wave tile: 32(M) x 64(N), 2x4 fragments of v_wmma_f32_16x16x32_bf16.
// Block = 128 threads = 4 waves covering 256 columns of N.
// TRANSC: store C[m*ldc + n] (final [C][N] output, residual = x in same layout).
// ---------------------------------------------------------------------------
template <bool RES, bool TRANSC>
__global__ __launch_bounds__(128) void gemm_wmma_bf16(
    const bf16_t* __restrict__ A, const bf16_t* __restrict__ B,
    const float* __restrict__ bias, const float* __restrict__ Res,
    float* __restrict__ C, int K, int lda, int ldb, int ldc)
{
  const int lane = threadIdx.x & 31;
  const int wave = threadIdx.x >> 5;
  const int n0 = blockIdx.x * 256 + wave * 64;
  const int m0 = blockIdx.y * 32;
  const int l16 = lane & 15;
  const int hi  = lane >> 4;

  v8f acc[2][4];
#pragma unroll
  for (int mi = 0; mi < 2; ++mi)
#pragma unroll
    for (int ni = 0; ni < 4; ++ni)
#pragma unroll
      for (int d = 0; d < 8; ++d) acc[mi][ni][d] = 0.0f;

  for (int k0 = 0; k0 < K; k0 += 32) {
    v16bf af[2];
#pragma unroll
    for (int mi = 0; mi < 2; ++mi)
      af[mi] = load_frag_a(A + (size_t)(m0 + mi * 16 + l16) * lda + k0 + hi * 8);
    v16bf bfr[4];
#pragma unroll
    for (int ni = 0; ni < 4; ++ni)
      bfr[ni] = load_frag_a(B + (size_t)(n0 + ni * 16 + l16) * ldb + k0 + hi * 8);
#pragma unroll
    for (int mi = 0; mi < 2; ++mi)
#pragma unroll
      for (int ni = 0; ni < 4; ++ni)
        acc[mi][ni] = __builtin_amdgcn_wmma_f32_16x16x32_bf16(
            false, af[mi], false, bfr[ni], (short)0, acc[mi][ni], false, false);
  }

  // D layout: lane -> n = l16 (+frag offset), m = hi*8 + d
#pragma unroll
  for (int mi = 0; mi < 2; ++mi) {
#pragma unroll
    for (int ni = 0; ni < 4; ++ni) {
      const int mbase = m0 + mi * 16 + hi * 8;
      const int nn = n0 + ni * 16 + l16;
      if constexpr (!TRANSC) {
        v8f r = acc[mi][ni];
        const v8f bv = *(const v8f*)(bias + mbase);
        if constexpr (RES) {
          const v8f rv = *(const v8f*)(Res + (size_t)nn * ldc + mbase);
#pragma unroll
          for (int d = 0; d < 8; ++d) r[d] += bv[d] + rv[d];
        } else {
#pragma unroll
          for (int d = 0; d < 8; ++d) r[d] += bv[d];
        }
        *(v8f*)(C + (size_t)nn * ldc + mbase) = r;
      } else {
        float* cp = C + (size_t)mbase * ldc + nn;
        const float* rp = RES ? (Res + (size_t)mbase * ldc + nn) : nullptr;
        const v8f bv = *(const v8f*)(bias + mbase);
#pragma unroll
        for (int d = 0; d < 8; ++d) {
          float v = acc[mi][ni][d] + bv[d];
          if constexpr (RES) v += rp[(size_t)d * ldc];
          cp[(size_t)d * ldc] = v;
        }
      }
    }
  }
}

// ---------------------------------------------------------------------------
// Utility kernels
// ---------------------------------------------------------------------------
__global__ void f32_to_bf16(const float* __restrict__ in, bf16_t* __restrict__ out, int n) {
  int i = blockIdx.x * blockDim.x + threadIdx.x;
  if (i < n) out[i] = (bf16_t)in[i];
}

// x [CIN][NSP] f32  ->  xT [NSP][CIN] bf16, LDS-tiled 32x32 (coalesced both ways)
__global__ __launch_bounds__(256) void transpose_to_bf16(
    const float* __restrict__ in, bf16_t* __restrict__ out, int R, int Ncols)
{
  __shared__ float tile[32][33];
  const int ntn = Ncols >> 5;
  const int c0 = (blockIdx.x / ntn) << 5;
  const int n0 = (blockIdx.x % ntn) << 5;
  const int tx = threadIdx.x & 31, ty = threadIdx.x >> 5;
#pragma unroll
  for (int i = 0; i < 4; ++i)
    tile[ty + 8 * i][tx] = in[(size_t)(c0 + ty + 8 * i) * Ncols + n0 + tx];
  __syncthreads();
#pragma unroll
  for (int i = 0; i < 4; ++i)
    out[(size_t)(n0 + 8 * i + ty) * R + c0 + tx] = (bf16_t)tile[tx][ty + 8 * i];
}

// conv = conv + dwconv3x3x3(bn1(conv)) + b_dw.  x1T [NSP][128], conv ch at +64.
__global__ __launch_bounds__(256) void dwconv_kernel(
    const float* __restrict__ x1T, const float* __restrict__ wdw,
    const float* __restrict__ bdw, const float* __restrict__ g,
    const float* __restrict__ b, const float* __restrict__ mn,
    const float* __restrict__ vr, float* __restrict__ convOut)
{
  const int idx = blockIdx.x * 256 + threadIdx.x;       // NSP*64
  const int c = idx & 63;
  const int n = idx >> 6;
  const int t = n >> 10, h = (n >> 5) & 31, w = n & 31;
  const float scale = g[c] * rsqrtf(vr[c] + EPSC);
  const float shift = b[c] - mn[c] * scale;
  const float center = x1T[(size_t)n * CDIM + CC + c];
  float acc = 0.0f;
  for (int dt = -1; dt <= 1; ++dt) {
    const int tt = t + dt;
    if (tt < 0 || tt > 7) continue;
    for (int dh = -1; dh <= 1; ++dh) {
      const int hh = h + dh;
      if (hh < 0 || hh > 31) continue;
      for (int dw2 = -1; dw2 <= 1; ++dw2) {
        const int ww = w + dw2;
        if (ww < 0 || ww > 31) continue;
        const float v = x1T[(size_t)((tt << 10) + (hh << 5) + ww) * CDIM + CC + c];
        acc += (v * scale + shift) * wdw[c * 27 + (dt + 1) * 9 + (dh + 1) * 3 + (dw2 + 1)];
      }
    }
  }
  convOut[idx] = center + acc + bdw[c];
}

// trilinear 2x downsample of SA slice: x1T[:, 0:64] (8,32,32) -> saD [ND][64]
__global__ __launch_bounds__(256) void downsample_kernel(
    const float* __restrict__ x1T, float* __restrict__ saD)
{
  const int idx = blockIdx.x * 256 + threadIdx.x;       // ND*64
  const int c = idx & 63;
  const int nd = idx >> 6;
  const int td = nd >> 8, hd = (nd >> 4) & 15, wd = nd & 15;
  float s = 0.0f;
#pragma unroll
  for (int a = 0; a < 2; ++a)
#pragma unroll
    for (int e = 0; e < 2; ++e)
#pragma unroll
      for (int f = 0; f < 2; ++f) {
        const int n = (((2 * td + a) << 10) + ((2 * hd + e) << 5) + (2 * wd + f));
        s += x1T[(size_t)n * CDIM + c];
      }
  saD[idx] = s * 0.125f;
}

// LayerNorm over 64 channels; one wave32 per row; output bf16.
__global__ __launch_bounds__(256) void ln64_kernel(
    const float* __restrict__ in, const float* __restrict__ g,
    const float* __restrict__ b, bf16_t* __restrict__ out, int rows)
{
  const int lane = threadIdx.x & 31;
  const int row = blockIdx.x * 8 + (threadIdx.x >> 5);
  if (row >= rows) return;
  const float* r = in + (size_t)row * 64;
  const float v0 = r[lane], v1 = r[lane + 32];
  float s = v0 + v1, sq = v0 * v0 + v1 * v1;
#pragma unroll
  for (int o = 16; o > 0; o >>= 1) {
    s  += __shfl_xor(s, o, 32);
    sq += __shfl_xor(sq, o, 32);
  }
  const float mean = s * (1.0f / 64.0f);
  const float var = sq * (1.0f / 64.0f) - mean * mean;
  const float rs = rsqrtf(var + EPSC);
  out[(size_t)row * 64 + lane]      = (bf16_t)((v0 - mean) * rs * g[lane] + b[lane]);
  out[(size_t)row * 64 + lane + 32] = (bf16_t)((v1 - mean) * rs * g[lane + 32] + b[lane + 32]);
}

// bn2(conv) -> bf16 [NSP][64]
__global__ __launch_bounds__(256) void bn_to_bf16(
    const float* __restrict__ in, const float* __restrict__ g,
    const float* __restrict__ b, const float* __restrict__ mn,
    const float* __restrict__ vr, bf16_t* __restrict__ out)
{
  const int idx = blockIdx.x * 256 + threadIdx.x;       // NSP*64
  const int c = idx & 63;
  const float scale = g[c] * rsqrtf(vr[c] + EPSC);
  out[idx] = (bf16_t)(in[idx] * scale + (b[c] - mn[c] * scale));
}

// Online-softmax attention (exp2 domain), d=8 per head, 8 heads.
// qkv rows: [q(64)|k(64)|v(64)]. K/V staged into LDS in 512-key chunks (32 KB),
// via GLOBAL_LOAD_ASYNC_TO_LDS_B128 + s_wait_asynccnt when available.
// Writes f32 (+resid) to outf and/or bf16 to outb (uniform pointer checks).
#define ACHUNK 512
__global__ __launch_bounds__(256) void attention_kernel(
    const float* __restrict__ Q, const float* __restrict__ KV,
    const float* __restrict__ resid, float* __restrict__ outf,
    bf16_t* __restrict__ outb, int Nk, int ldq, int ldkv)
{
  __shared__ float kb[ACHUNK * 8];
  __shared__ float vb[ACHUNK * 8];
  const int h = blockIdx.y;
  const int qi = blockIdx.x * 256 + threadIdx.x;
  float q[8];
  const float* qp = Q + (size_t)qi * ldq + h * 8;
#pragma unroll
  for (int d = 0; d < 8; ++d) q[d] = qp[d] * QSCALE;  // fold scale & log2e
  float mM = -3.0e38f, ss = 0.0f;
  float acc[8] = {0, 0, 0, 0, 0, 0, 0, 0};
  for (int c0 = 0; c0 < Nk; c0 += ACHUNK) {
    __syncthreads();
#if USE_ASYNC_LDS
    for (int i = threadIdx.x; i < ACHUNK * 2; i += 256) {
      const int rrow = i >> 1, half = (i & 1) * 4;
      const float* base = KV + (size_t)(c0 + rrow) * ldkv + h * 8;
      __builtin_amdgcn_global_load_async_to_lds_b128(
          (v4i*)(base + 64 + half), (v4i*)(kb + rrow * 8 + half), 0, 0);
      __builtin_amdgcn_global_load_async_to_lds_b128(
          (v4i*)(base + 128 + half), (v4i*)(vb + rrow * 8 + half), 0, 0);
    }
    __builtin_amdgcn_s_wait_asynccnt(0);
#else
    for (int i = threadIdx.x; i < ACHUNK * 8; i += 256) {
      const int rrow = i >> 3, d = i & 7;
      const float* kvp = KV + (size_t)(c0 + rrow) * ldkv + h * 8;
      kb[i] = kvp[64 + d];
      vb[i] = kvp[128 + d];
    }
#endif
    __syncthreads();
    for (int j = 0; j < ACHUNK; ++j) {
      const float* kr = kb + j * 8;
      float sc = 0.0f;
#pragma unroll
      for (int d = 0; d < 8; ++d) sc = fmaf(q[d], kr[d], sc);
      const float mn2 = fmaxf(mM, sc);
      const float corr = exp2f(mM - mn2);
      const float p = exp2f(sc - mn2);
      ss = ss * corr + p;
      const float* vr2 = vb + j * 8;
#pragma unroll
      for (int d = 0; d < 8; ++d) acc[d] = fmaf(acc[d], corr, p * vr2[d]);
      mM = mn2;
    }
  }
  const float inv = 1.0f / ss;
  if (outf) {
    float* op = outf + (size_t)qi * 64 + h * 8;
    const float* rp = resid ? resid + (size_t)qi * 64 + h * 8 : nullptr;
#pragma unroll
    for (int d = 0; d < 8; ++d) op[d] = acc[d] * inv + (rp ? rp[d] : 0.0f);
  }
  if (outb) {
    bf16_t* ob = outb + (size_t)qi * 64 + h * 8;
#pragma unroll
    for (int d = 0; d < 8; ++d) ob[d] = (bf16_t)(acc[d] * inv);
  }
}

// half-pixel linear sample helper (jax.image.resize convention, edge clamp)
__device__ inline void lin_coord(int i, int outd, int ind, int& i0, int& i1, float& f) {
  const float c = (i + 0.5f) * ((float)ind / (float)outd) - 0.5f;
  const float fl = floorf(c);
  f = c - fl;
  int ii = (int)fl;
  i0 = ii < 0 ? 0 : ii;
  i1 = (ii + 1 > ind - 1) ? ind - 1 : ii + 1;
}

// outT[n][0:64]=bf16(convFinal), outT[n][64:128]=bf16(trilinear-up(saFinal))
__global__ __launch_bounds__(256) void combine_kernel(
    const float* __restrict__ convF, const float* __restrict__ saF,
    bf16_t* __restrict__ outT)
{
  const int idx = blockIdx.x * 256 + threadIdx.x;       // NSP*128
  const int c = idx & 127;
  const int n = idx >> 7;
  float v;
  if (c < 64) {
    v = convF[(size_t)n * 64 + c];
  } else {
    const int c2 = c - 64;
    const int t = n >> 10, hh = (n >> 5) & 31, ww = n & 31;
    int t0, t1, h0, h1, w0, w1;
    float ft, fh, fw;
    lin_coord(t, 8, 4, t0, t1, ft);
    lin_coord(hh, 32, 16, h0, h1, fh);
    lin_coord(ww, 32, 16, w0, w1, fw);
    auto S = [&](int a, int e, int f) -> float {
      return saF[(size_t)((a << 8) + (e << 4) + f) * 64 + c2];
    };
    const float v00 = S(t0, h0, w0) * (1.0f - fw) + S(t0, h0, w1) * fw;
    const float v01 = S(t0, h1, w0) * (1.0f - fw) + S(t0, h1, w1) * fw;
    const float v10 = S(t1, h0, w0) * (1.0f - fw) + S(t1, h0, w1) * fw;
    const float v11 = S(t1, h1, w0) * (1.0f - fw) + S(t1, h1, w1) * fw;
    const float va = v00 * (1.0f - fh) + v01 * fh;
    const float vbq = v10 * (1.0f - fh) + v11 * fh;
    v = va * (1.0f - ft) + vbq * ft;
  }
  outT[idx] = (bf16_t)v;
}

// ---------------------------------------------------------------------------
// Host orchestration
// ---------------------------------------------------------------------------
extern "C" void kernel_launch(void* const* d_in, const int* in_sizes, int n_in,
                              void* d_out, int out_size, void* d_ws, size_t ws_size,
                              hipStream_t stream) {
  (void)in_sizes; (void)n_in; (void)out_size; (void)ws_size;
  const float* x       = (const float*)d_in[0];
  const float* w_conv1 = (const float*)d_in[1];
  const float* b_conv1 = (const float*)d_in[2];
  const float* bn1_g   = (const float*)d_in[3];
  const float* bn1_b   = (const float*)d_in[4];
  const float* bn1_m   = (const float*)d_in[5];
  const float* bn1_v   = (const float*)d_in[6];
  const float* w_dw    = (const float*)d_in[7];
  const float* b_dw    = (const float*)d_in[8];
  const float* ln1_g   = (const float*)d_in[9];
  const float* ln1_b   = (const float*)d_in[10];
  const float* w_up    = (const float*)d_in[11];
  const float* b_up    = (const float*)d_in[12];
  const float* bn2_g   = (const float*)d_in[13];
  const float* bn2_b   = (const float*)d_in[14];
  const float* bn2_m   = (const float*)d_in[15];
  const float* bn2_v   = (const float*)d_in[16];
  const float* w_ccc   = (const float*)d_in[17];
  const float* b_ccc   = (const float*)d_in[18];
  const float* ln2_g   = (const float*)d_in[19];
  const float* ln2_b   = (const float*)d_in[20];
  const float* w_ccs   = (const float*)d_in[21];
  const float* b_ccs   = (const float*)d_in[22];
  const float* w_fc    = (const float*)d_in[23];
  const float* b_fc    = (const float*)d_in[24];
  const float* w_fs    = (const float*)d_in[25];
  const float* b_fs    = (const float*)d_in[26];
  const float* w_conv2 = (const float*)d_in[27];
  const float* b_conv2 = (const float*)d_in[28];
  float* out = (float*)d_out;

  char* wsp = (char*)d_ws;
  auto take = [&](size_t bytes) -> void* { void* p = (void*)wsp; wsp += bytes; return p; };

  bf16_t* xT_bf    = (bf16_t*)take((size_t)NSP * CIN * 2);   // 16 MB
  bf16_t* w1_bf    = (bf16_t*)take((size_t)CDIM * CIN * 2);
  bf16_t* wup_bf   = (bf16_t*)take((size_t)QKV3 * CC * 2);
  bf16_t* wccc_bf  = (bf16_t*)take((size_t)QKV3 * CC * 2);
  bf16_t* wccs_bf  = (bf16_t*)take((size_t)QKV3 * CC * 2);
  bf16_t* wfc_bf   = (bf16_t*)take((size_t)CC * CC * 2);
  bf16_t* wfs_bf   = (bf16_t*)take((size_t)CC * CC * 2);
  bf16_t* w2_bf    = (bf16_t*)take((size_t)CIN * CDIM * 2);
  float*  x1T      = (float*)take((size_t)NSP * CDIM * 4);   // 4 MB
  float*  convBuf  = (float*)take((size_t)NSP * CC * 4);     // 2 MB
  float*  saD      = (float*)take((size_t)ND * CC * 4);
  bf16_t* saLN_bf  = (bf16_t*)take((size_t)ND * CC * 2);
  float*  sa3T     = (float*)take((size_t)ND * QKV3 * 4);
  float*  saAttn   = (float*)take((size_t)ND * CC * 4);
  bf16_t* bn2c_bf  = (bf16_t*)take((size_t)NSP * CC * 2);
  float*  convqkvT = (float*)take((size_t)NSP * QKV3 * 4);   // 6 MB
  bf16_t* saLN2_bf = (bf16_t*)take((size_t)ND * CC * 2);
  float*  saqkvT   = (float*)take((size_t)ND * QKV3 * 4);
  bf16_t* attnA_bf = (bf16_t*)take((size_t)NSP * CC * 2);    // 1 MB
  float*  convFin  = (float*)take((size_t)NSP * CC * 4);     // 2 MB
  bf16_t* attnB_bf = (bf16_t*)take((size_t)ND * CC * 2);
  float*  saFin    = (float*)take((size_t)ND * CC * 4);
  bf16_t* outT_bf  = (bf16_t*)take((size_t)NSP * CDIM * 2);  // 2 MB

  // --- weight conversions to bf16 (row-major [M][K] kept) ---
  auto cvt = [&](const float* src, bf16_t* dst, int n) {
    f32_to_bf16<<<(n + 255) / 256, 256, 0, stream>>>(src, dst, n);
  };
  cvt(w_conv1, w1_bf, CDIM * CIN);
  cvt(w_up, wup_bf, QKV3 * CC);
  cvt(w_ccc, wccc_bf, QKV3 * CC);
  cvt(w_ccs, wccs_bf, QKV3 * CC);
  cvt(w_fc, wfc_bf, CC * CC);
  cvt(w_fs, wfs_bf, CC * CC);
  cvt(w_conv2, w2_bf, CIN * CDIM);

  // --- x [1024][8192] -> xT bf16 [8192][1024] ---
  transpose_to_bf16<<<(CIN / 32) * (NSP / 32), 256, 0, stream>>>(x, xT_bf, CIN, NSP);

  // --- GEMM1: x1T [8192][128] = xT * w_conv1^T + b_conv1 ---
  gemm_wmma_bf16<false, false><<<dim3(NSP / 256, CDIM / 32), 128, 0, stream>>>(
      w1_bf, xT_bf, b_conv1, nullptr, x1T, CIN, CIN, CIN, CDIM);

  // --- depthwise conv path: conv = conv + dw(bn1(conv)) + b_dw ---
  dwconv_kernel<<<(NSP * CC) / 256, 256, 0, stream>>>(
      x1T, w_dw, b_dw, bn1_g, bn1_b, bn1_m, bn1_v, convBuf);

  // --- SA path: downsample qkv slice to saD [1024][64] ---
  downsample_kernel<<<(ND * CC) / 256, 256, 0, stream>>>(x1T, saD);

  // --- LN1 + up-projection: sa3T [1024][192] ---
  ln64_kernel<<<ND / 8, 256, 0, stream>>>(saD, ln1_g, ln1_b, saLN_bf, ND);
  gemm_wmma_bf16<false, false><<<dim3(ND / 256, QKV3 / 32), 128, 0, stream>>>(
      wup_bf, saLN_bf, b_up, nullptr, sa3T, CC, CC, CC, QKV3);

  // --- self-attention (1024 tokens), residual saD -> saAttn (f32) ---
  attention_kernel<<<dim3(ND / 256, 8), 256, 0, stream>>>(
      sa3T, sa3T, saD, saAttn, nullptr, ND, QKV3, QKV3);

  // --- conv_qkv = conv1x1(bn2(conv), w_ccc) -> convqkvT [8192][192] ---
  bn_to_bf16<<<(NSP * CC) / 256, 256, 0, stream>>>(
      convBuf, bn2_g, bn2_b, bn2_m, bn2_v, bn2c_bf);
  gemm_wmma_bf16<false, false><<<dim3(NSP / 256, QKV3 / 32), 128, 0, stream>>>(
      wccc_bf, bn2c_bf, b_ccc, nullptr, convqkvT, CC, CC, CC, QKV3);

  // --- sa_qkv = ln2(sa) @ w_ccs^T -> saqkvT [1024][192] ---
  ln64_kernel<<<ND / 8, 256, 0, stream>>>(saAttn, ln2_g, ln2_b, saLN2_bf, ND);
  gemm_wmma_bf16<false, false><<<dim3(ND / 256, QKV3 / 32), 128, 0, stream>>>(
      wccs_bf, saLN2_bf, b_ccs, nullptr, saqkvT, CC, CC, CC, QKV3);

  // --- cross-attn A: attend(cq, sk, sv) -> attnA bf16; fc GEMM + residual conv ---
  attention_kernel<<<dim3(NSP / 256, 8), 256, 0, stream>>>(
      convqkvT, saqkvT, nullptr, nullptr, attnA_bf, ND, QKV3, QKV3);
  gemm_wmma_bf16<true, false><<<dim3(NSP / 256, CC / 32), 128, 0, stream>>>(
      wfc_bf, attnA_bf, b_fc, convBuf, convFin, CC, CC, CC, CC);

  // --- cross-attn B: attend(sq, ck, cv) -> attnB bf16; fs GEMM + residual sa ---
  attention_kernel<<<dim3(ND / 256, 8), 256, 0, stream>>>(
      saqkvT, convqkvT, nullptr, nullptr, attnB_bf, NSP, QKV3, QKV3);
  gemm_wmma_bf16<true, false><<<dim3(ND / 256, CC / 32), 128, 0, stream>>>(
      wfs_bf, attnB_bf, b_fs, saAttn, saFin, CC, CC, CC, CC);

  // --- concat [conv | up(sa)] -> outT bf16 [8192][128] ---
  combine_kernel<<<(NSP * CDIM) / 256, 256, 0, stream>>>(convFin, saFin, outT_bf);

  // --- final: out[C][N] = x + conv1x1(outT, w_conv2) + b_conv2 (transposed store) ---
  gemm_wmma_bf16<true, true><<<dim3(NSP / 256, CIN / 32), 128, 0, stream>>>(
      w2_bf, outT_bf, b_conv2, x, out, CDIM, CDIM, CDIM, NSP);
}
